// GQA_68564857913784
// MI455X (gfx1250) — matrix operand order repaired and checked by reading, not compiled, not run on hardware
//
#include <hip/hip_runtime.h>
#include <math.h>

typedef __bf16 bf16_t;
typedef bf16_t v16bf __attribute__((ext_vector_type(16)));
typedef float  v8f   __attribute__((ext_vector_type(8)));

union Frag {
  v16bf v;
  unsigned int w[8];
};

// Problem constants
constexpr int BB  = 4;
constexpr int SS  = 2048;
constexpr int DD  = 2048;
constexpr int HH  = 16;
constexpr int KVH = 4;
constexpr int HD  = 128;
constexpr int NREP = HH / KVH;
constexpr float SCALE = 0.08838834764831845f; // 1/sqrt(128)

// ---------------------------------------------------------------------------
// CDNA5 async memory->LDS helpers (ASYNCcnt-tracked DMA, gfx1250)
// ---------------------------------------------------------------------------
__device__ __forceinline__ void async_b128(unsigned lds_off, const void* g) {
  // GLOBAL_LOAD_ASYNC_TO_LDS_B128, GV mode (64-bit address in VGPR pair).
  asm volatile("global_load_async_to_lds_b128 %0, %1, off"
               :: "v"(lds_off), "v"(g) : "memory");
}
__device__ __forceinline__ void async_wait0() {
  asm volatile("s_wait_asynccnt 0x0" ::: "memory");
}
__device__ __forceinline__ void async_wait4() {
  asm volatile("s_wait_asynccnt 0x4" ::: "memory");
}

// ---------------------------------------------------------------------------
// bf16 WMMA GEMM with bias, async double-buffered LDS pipeline.
//   C[M,N] = Ab[M,K](bf16) @ Wt[N,K](bf16, pre-transposed)^T + bias[N]
// Block tile 128x128, 8 waves (2x4), each wave 64x32 = 4x2 WMMA tiles.
// ---------------------------------------------------------------------------
#define BM 128
#define BN 128
#define BK 32

__global__ __launch_bounds__(256) void gemm_bf16_async(
    const bf16_t* __restrict__ Ab, const bf16_t* __restrict__ Wt,
    const float* __restrict__ bias, float* __restrict__ C, int N, int K)
{
  __shared__ __align__(16) bf16_t As[2][BM][BK];   // [m][k]
  __shared__ __align__(16) bf16_t Bs[2][BN][BK];   // [n][k]

  const int tid  = threadIdx.x;
  const int lane = tid & 31;
  const int wv   = tid >> 5;        // 0..7
  const int wr   = wv >> 2;         // 0..1 -> 64-row band
  const int wc   = wv & 3;          // 0..3 -> 32-col band
  const int ln   = lane & 15;
  const int hl   = lane >> 4;
  const int row0 = blockIdx.y * BM;
  const int col0 = blockIdx.x * BN;

  // copy assignment: thread -> (row cr, 16-element chunk ch); 2x b128 each for A and B
  const int cr = tid >> 1;
  const int ch = (tid & 1) * 16;
  const bf16_t* ga = Ab + (size_t)(row0 + cr) * K + ch;
  const bf16_t* gb = Wt + (size_t)(col0 + cr) * K + ch;

  auto issue_copy = [&](int buf, int k0) {
    const unsigned da = (unsigned)(size_t)&As[buf][cr][ch];
    const unsigned db = (unsigned)(size_t)&Bs[buf][cr][ch];
    async_b128(da,      ga + k0);
    async_b128(da + 16, ga + k0 + 8);
    async_b128(db,      gb + k0);
    async_b128(db + 16, gb + k0 + 8);
  };

  v8f acc[4][2];
#pragma unroll
  for (int r = 0; r < 4; ++r)
#pragma unroll
    for (int c = 0; c < 2; ++c)
#pragma unroll
      for (int j = 0; j < 8; ++j) acc[r][c][j] = 0.0f;

  issue_copy(0, 0);

  int buf = 0;
  for (int k0 = 0; k0 < K; k0 += BK, buf ^= 1) {
    if (k0 + BK < K) {
      issue_copy(buf ^ 1, k0 + BK);  // prefetch next tile while current computes
      async_wait4();                  // current buffer's 4 DMAs are done
    } else {
      async_wait0();
    }
    __syncthreads();

    Frag af[4], bfr[2];
#pragma unroll
    for (int r = 0; r < 4; ++r) {
      const int m = wr * 64 + r * 16 + ln;
#pragma unroll
      for (int p = 0; p < 8; ++p) {
        const int kk = 16 * hl + 2 * p;  // A-frag K map: k = 16*hl + i, contiguous
        af[r].w[p] = *reinterpret_cast<const unsigned int*>(&As[buf][m][kk]);
      }
    }
#pragma unroll
    for (int c = 0; c < 2; ++c) {
      const int n = wc * 32 + c * 16 + ln;
#pragma unroll
      for (int p = 0; p < 8; ++p)   // B-frag: k = 16*hl + i, contiguous
        bfr[c].w[p] = *reinterpret_cast<const unsigned int*>(&Bs[buf][n][16 * hl + 2 * p]);
    }
#pragma unroll
    for (int r = 0; r < 4; ++r)
#pragma unroll
      for (int c = 0; c < 2; ++c)
        acc[r][c] = __builtin_amdgcn_wmma_f32_16x16x32_bf16(
            false, af[r].v, false, bfr[c].v, (short)0, acc[r][c], false, false);
    __syncthreads();
  }

  // Epilogue: lane holds col n = ln, rows m = j + 8*hl (C/D layout).
#pragma unroll
  for (int r = 0; r < 4; ++r)
#pragma unroll
    for (int c = 0; c < 2; ++c) {
      const int col = col0 + wc * 32 + c * 16 + ln;
      const float bvv = bias[col];
#pragma unroll
      for (int j = 0; j < 8; ++j) {
        const int rowg = row0 + wr * 64 + r * 16 + j + 8 * hl;
        C[(size_t)rowg * N + col] = acc[r][c][j] + bvv;
      }
    }
}

// ---------------------------------------------------------------------------
// One-time conversions: f32 -> bf16 (x), and weight K x N f32 -> N x K bf16.
// ---------------------------------------------------------------------------
__global__ __launch_bounds__(256) void f2b_kernel(
    const float* __restrict__ in, bf16_t* __restrict__ out, int total)
{
  const int idx = blockIdx.x * blockDim.x + threadIdx.x;
  if (idx < total) out[idx] = (bf16_t)in[idx];
}

__global__ __launch_bounds__(256) void wtrans_kernel(
    const float* __restrict__ W, bf16_t* __restrict__ Wt, int N, int K, int total)
{
  const int idx = blockIdx.x * blockDim.x + threadIdx.x;
  if (idx >= total) return;
  const int n = idx % N;
  const int k = idx / N;
  Wt[(size_t)n * K + k] = (bf16_t)W[idx];
}

// ---------------------------------------------------------------------------
// RoPE (interleaved pairs) + transpose to [b][h][s][d], f32 -> bf16.
// ---------------------------------------------------------------------------
__global__ __launch_bounds__(256) void rope_kernel(
    const float* __restrict__ in, bf16_t* __restrict__ out,
    const float* __restrict__ cosb, const float* __restrict__ sinb,
    int nh, int total)
{
  const int idx = blockIdx.x * blockDim.x + threadIdx.x;
  if (idx >= total) return;
  constexpr int HP = HD / 2;
  int d2 = idx % HP;
  int t  = idx / HP;
  const int hh = t % nh; t /= nh;
  const int s  = t % SS;
  const int b  = t / SS;
  const size_t inoff = ((size_t)(b * SS + s) * nh + hh) * HD + 2 * d2;
  const float xr = in[inoff], xi = in[inoff + 1];
  const float c = cosb[s * HP + d2], sn = sinb[s * HP + d2];
  const float yr = xr * c - xi * sn;
  const float yi = xr * sn + xi * c;
  const size_t oo = (((size_t)(b * nh + hh)) * SS + s) * HD + 2 * d2;
  out[oo]     = (bf16_t)yr;
  out[oo + 1] = (bf16_t)yi;
}

// V: (b,s,kvh,d) f32 -> (b,kvh,d,s) bf16 (so P@V B-frags load contiguously)
__global__ __launch_bounds__(256) void vtrans_kernel(
    const float* __restrict__ in, bf16_t* __restrict__ out, int total)
{
  const int idx = blockIdx.x * blockDim.x + threadIdx.x;
  if (idx >= total) return;
  const int d = idx % HD;
  int t = idx / HD;
  const int kvh = t % KVH; t /= KVH;
  const int s = t % SS;
  const int b = t / SS;
  out[(((size_t)(b * KVH + kvh)) * HD + d) * SS + s] = (bf16_t)in[idx];
}

// ---------------------------------------------------------------------------
// Flash attention: one wave per 16-row query tile, key blocks of 32,
// online softmax, bf16 WMMA for Q.K^T and P.V. Output bf16 (B,S,D).
// ---------------------------------------------------------------------------
__global__ __launch_bounds__(128) void attn_kernel(
    const bf16_t* __restrict__ Q,   // [b][h][s][d]
    const bf16_t* __restrict__ Kc,  // [b][kvh][s][d] (+32 rows pad)
    const bf16_t* __restrict__ Vt,  // [b][kvh][d][s] (+pad)
    bf16_t* __restrict__ O)         // [b][s][h*HD]  (B,S,D) bf16
{
  __shared__ bf16_t Pl[4][16][32];  // per-wave P transpose scratch

  const int wv   = threadIdx.x >> 5;
  const int lane = threadIdx.x & 31;
  const int ln = lane & 15, hl = lane >> 4;
  const int qt = blockIdx.x * 4 + wv;
  const int h  = blockIdx.y;
  const int b  = blockIdx.z;
  const int kvh = h / NREP;
  const int m0 = qt * 16;

  const bf16_t* qbase = Q  + ((size_t)(b * HH + h) * SS + m0) * HD;
  const bf16_t* kbase = Kc + ((size_t)(b * KVH + kvh)) * SS * HD;
  const bf16_t* vbase = Vt + ((size_t)(b * KVH + kvh)) * (size_t)HD * SS;

  // Q fragments: 4 chunks of K-dim 32. A-frag row m = ln.
  Frag qf[4];
#pragma unroll
  for (int dc = 0; dc < 4; ++dc)
#pragma unroll
    for (int p = 0; p < 8; ++p) {
      const int kk = dc * 32 + 16 * hl + 2 * p;
      qf[dc].w[p] = *reinterpret_cast<const unsigned int*>(&qbase[(size_t)ln * HD + kk]);
    }

  v8f Oacc[8];
  float rmax[8], rsum[8];
#pragma unroll
  for (int nc = 0; nc < 8; ++nc)
#pragma unroll
    for (int j = 0; j < 8; ++j) Oacc[nc][j] = 0.0f;
#pragma unroll
  for (int j = 0; j < 8; ++j) { rmax[j] = -1e30f; rsum[j] = 0.0f; }

  for (int t0 = 0; t0 <= m0 + 15; t0 += 32) {
    // ---- scores: two 16x16 tiles over this 32-key block ----
    v8f sc[2];
#pragma unroll
    for (int tt = 0; tt < 2; ++tt) {
#pragma unroll
      for (int j = 0; j < 8; ++j) sc[tt][j] = 0.0f;
#pragma unroll
      for (int dc = 0; dc < 4; ++dc) {
        Frag kf;  // B[d][t] = K[t][d]; element i -> d = dc*32 + 16*hl + i (contig)
        const bf16_t* kp = &kbase[(size_t)(t0 + tt * 16 + ln) * HD + dc * 32 + hl * 16];
#pragma unroll
        for (int p = 0; p < 8; ++p)
          kf.w[p] = reinterpret_cast<const unsigned int*>(kp)[p];
        sc[tt] = __builtin_amdgcn_wmma_f32_16x16x32_bf16(
            false, qf[dc].v, false, kf.v, (short)0, sc[tt], false, false);
      }
    }
    // ---- scale + causal mask ----
#pragma unroll
    for (int tt = 0; tt < 2; ++tt) {
      const int tg = t0 + tt * 16 + ln;
#pragma unroll
      for (int j = 0; j < 8; ++j) {
        const int mg = m0 + j + 8 * hl;
        sc[tt][j] = sc[tt][j] * SCALE + ((tg > mg) ? -1e9f : 0.0f);
      }
    }
    // ---- row max over 32 keys (reduce across 16 lanes in each half) ----
    float tmax[8];
#pragma unroll
    for (int j = 0; j < 8; ++j) tmax[j] = fmaxf(sc[0][j], sc[1][j]);
#pragma unroll
    for (int d = 1; d < 16; d <<= 1)
#pragma unroll
      for (int j = 0; j < 8; ++j) tmax[j] = fmaxf(tmax[j], __shfl_xor(tmax[j], d, 32));

    float corr[8], mnew[8];
#pragma unroll
    for (int j = 0; j < 8; ++j) {
      mnew[j] = fmaxf(rmax[j], tmax[j]);
      corr[j] = __expf(rmax[j] - mnew[j]);
      rmax[j] = mnew[j];
    }
    // ---- P = exp(sc - mnew): stash to LDS in (m, t) layout; track row sums ----
    float psum[8];
#pragma unroll
    for (int j = 0; j < 8; ++j) psum[j] = 0.0f;
#pragma unroll
    for (int tt = 0; tt < 2; ++tt)
#pragma unroll
      for (int j = 0; j < 8; ++j) {
        const float p = __expf(sc[tt][j] - mnew[j]);
        psum[j] += p;
        Pl[wv][j + 8 * hl][tt * 16 + ln] = (bf16_t)p;
      }
    asm volatile("s_wait_dscnt 0" ::: "memory");  // DS in-order per wave; stop reordering
#pragma unroll
    for (int d = 1; d < 16; d <<= 1)
#pragma unroll
      for (int j = 0; j < 8; ++j) psum[j] += __shfl_xor(psum[j], d, 32);
#pragma unroll
    for (int j = 0; j < 8; ++j) rsum[j] = rsum[j] * corr[j] + psum[j];
    // ---- rescale O ----
#pragma unroll
    for (int nc = 0; nc < 8; ++nc)
#pragma unroll
      for (int j = 0; j < 8; ++j) Oacc[nc][j] *= corr[j];
    // ---- read P back as one 16x32 A-fragment ----
    Frag pf;
#pragma unroll
    for (int p = 0; p < 8; ++p) {
      const int kk = 16 * hl + 2 * p;
      pf.w[p] = *reinterpret_cast<const unsigned int*>(&Pl[wv][ln][kk]);
    }
    // ---- O += P @ V : 8 column chunks of HD ----
#pragma unroll
    for (int nc = 0; nc < 8; ++nc) {
      Frag vf;  // B[t][d]; Vt layout [d][s]: element i -> t = t0 + 16*hl + i (contig)
      const bf16_t* vp = &vbase[(size_t)(nc * 16 + ln) * SS + t0 + 16 * hl];
#pragma unroll
      for (int p = 0; p < 8; ++p)
        vf.w[p] = reinterpret_cast<const unsigned int*>(vp)[p];
      Oacc[nc] = __builtin_amdgcn_wmma_f32_16x16x32_bf16(
          false, pf.v, false, vf.v, (short)0, Oacc[nc], false, false);
    }
  }

  // ---- normalize and write (B,S,D) bf16 ----
  bf16_t* obase = O + ((size_t)b * SS + m0) * DD + h * HD;
  float inv[8];
#pragma unroll
  for (int j = 0; j < 8; ++j) inv[j] = 1.0f / rsum[j];
#pragma unroll
  for (int nc = 0; nc < 8; ++nc)
#pragma unroll
    for (int j = 0; j < 8; ++j)
      obase[(size_t)(j + 8 * hl) * DD + nc * 16 + ln] = (bf16_t)(Oacc[nc][j] * inv[j]);
}

// ---------------------------------------------------------------------------
extern "C" void kernel_launch(void* const* d_in, const int* in_sizes, int n_in,
                              void* d_out, int out_size, void* d_ws, size_t ws_size,
                              hipStream_t stream) {
  const float* x  = (const float*)d_in[0];
  const float* Wq = (const float*)d_in[1];
  const float* bq = (const float*)d_in[2];
  const float* Wk = (const float*)d_in[3];
  const float* bk = (const float*)d_in[4];
  const float* Wv = (const float*)d_in[5];
  const float* bv = (const float*)d_in[6];
  const float* Wo = (const float*)d_in[7];
  const float* bo = (const float*)d_in[8];
  const float* fcos = (const float*)d_in[9];
  const float* fsin = (const float*)d_in[10];
  float* out = (float*)d_out;

  const int M = BB * SS;  // 8192
  const int NKV = KVH * HD; // 512
  char* ws = (char*)d_ws;

  // workspace carve-up (bytes)
  size_t off = 0;
  auto carve = [&](size_t bytes) { char* p = ws + off; off += (bytes + 255) & ~(size_t)255; return p; };
  bf16_t* xb   = (bf16_t*)carve((size_t)M * DD * 2);          // 32 MB
  bf16_t* WqT  = (bf16_t*)carve((size_t)DD * DD * 2);         //  8 MB
  bf16_t* WkT  = (bf16_t*)carve((size_t)NKV * DD * 2);        //  2 MB
  bf16_t* WvT  = (bf16_t*)carve((size_t)NKV * DD * 2);        //  2 MB
  bf16_t* WoT  = (bf16_t*)carve((size_t)DD * DD * 2);         //  8 MB
  float*  qf32 = (float*)carve((size_t)M * DD * 4);           // 64 MB
  float*  kf32 = (float*)carve((size_t)M * NKV * 4);          // 16 MB
  float*  vf32 = (float*)carve((size_t)M * NKV * 4);          // 16 MB
  bf16_t* qb   = (bf16_t*)carve((size_t)M * DD * 2);          // 32 MB
  bf16_t* kb   = (bf16_t*)carve(((size_t)BB * KVH * SS + 32) * HD * 2);
  bf16_t* vtb  = (bf16_t*)carve(((size_t)BB * KVH * SS + 32) * HD * 2);
  bf16_t* attb = (bf16_t*)carve((size_t)M * DD * 2);          // 32 MB

  // 0) one-time precision/layout conversions
  {
    int t0 = M * DD;
    f2b_kernel<<<(t0 + 255) / 256, 256, 0, stream>>>(x, xb, t0);
    int tw = DD * DD;
    wtrans_kernel<<<(tw + 255) / 256, 256, 0, stream>>>(Wq, WqT, DD, DD, tw);
    wtrans_kernel<<<(tw + 255) / 256, 256, 0, stream>>>(Wo, WoT, DD, DD, tw);
    int tk = DD * NKV;
    wtrans_kernel<<<(tk + 255) / 256, 256, 0, stream>>>(Wk, WkT, NKV, DD, tk);
    wtrans_kernel<<<(tk + 255) / 256, 256, 0, stream>>>(Wv, WvT, NKV, DD, tk);
  }

  // 1) QKV projections (async-pipelined bf16 WMMA, f32 out)
  gemm_bf16_async<<<dim3(DD / BN, M / BM), 256, 0, stream>>>(xb, WqT, bq, qf32, DD, DD);
  gemm_bf16_async<<<dim3(NKV / BN, M / BM), 256, 0, stream>>>(xb, WkT, bk, kf32, NKV, DD);
  gemm_bf16_async<<<dim3(NKV / BN, M / BM), 256, 0, stream>>>(xb, WvT, bv, vf32, NKV, DD);

  // 2) RoPE + transpose to head-major bf16; V -> [b][kvh][d][s]
  {
    int tq = BB * SS * HH * (HD / 2);
    rope_kernel<<<(tq + 255) / 256, 256, 0, stream>>>(qf32, qb, fcos, fsin, HH, tq);
    int tk = BB * SS * KVH * (HD / 2);
    rope_kernel<<<(tk + 255) / 256, 256, 0, stream>>>(kf32, kb, fcos, fsin, KVH, tk);
    int tv = BB * SS * KVH * HD;
    vtrans_kernel<<<(tv + 255) / 256, 256, 0, stream>>>(vf32, vtb, tv);
  }

  // 3) flash attention -> (B,S,D) bf16
  attn_kernel<<<dim3(SS / 16 / 4, HH, BB), 128, 0, stream>>>(qb, kb, vtb, attb);

  // 4) output projection (bf16 WMMA) -> f32 d_out
  gemm_bf16_async<<<dim3(DD / BN, M / BM), 256, 0, stream>>>(attb, WoT, bo, out, DD, DD);
}
